// EnhancedReprogrammingLayer_54949811585652
// MI455X (gfx1250) — matrix operand order repaired and checked by reading, not compiled, not run on hardware
//
#include <hip/hip_runtime.h>

// ---------------------------------------------------------------------------
// MI455X (gfx1250) implementation. wave32, WMMA f32_16x16x32_f16.
//
// Roofline: ~136 GFLOP total, ~200 MB mandatory HBM traffic (~9us @ 23.3TB/s)
// => compute bound => everything runs on v_wmma_f32_16x16x32_f16 with f32
// accumulation. Scores are never materialized to HBM (flash-style online
// softmax), Q/K/V/attn-out staged as f16 in workspace. The attention K tile
// is copied global->LDS with GLOBAL_LOAD_ASYNC_TO_LDS_B128 (ASYNCcnt path),
// bypassing VGPRs in the WMMA-dense inner loop.
// ---------------------------------------------------------------------------

typedef __attribute__((ext_vector_type(16))) _Float16 v16h;
typedef __attribute__((ext_vector_type(8)))  _Float16 v8h;
typedef __attribute__((ext_vector_type(8)))  float    v8f;

union V16 { v16h v; v8h h[2]; };
union H16 { _Float16 e[16]; v8h h[2]; };

__device__ __forceinline__ v8f wmma_f16f32(v16h a, v16h b, v8f c) {
  // D(16x16 f32) = A(16x32 f16) * B(32x16 f16) + C
  return __builtin_amdgcn_wmma_f32_16x16x32_f16(
      /*neg_a=*/false, a, /*neg_b=*/false, b,
      /*c_mod=*/(short)0, c, /*reuse_a=*/false, /*reuse_b=*/false);
}

// ---------------------------------------------------------------------------
// Tiled GEMM with torch-Linear semantics:  C[M,N] = A[M,K] * W[N,K]^T + bias
// Workgroup: 256 threads (8 waves), tile 128(M) x 128(N), K-step 32.
// Wave (wid&3, wid>>2) owns a 32x64 sub-tile = 2x4 WMMA tiles.
// InT:  float (convert to f16 on load) or _Float16 (direct).
// OutT: _Float16 (to workspace) or float (final output).
// Rows in [M, Mpad) are stored as zeros (used to pad S=1000 -> 1024).
// ---------------------------------------------------------------------------
template <typename InT, typename OutT>
__global__ __launch_bounds__(256)
void gemm_bias_kernel(const InT* __restrict__ A, const float* __restrict__ W,
                      const float* __restrict__ bias, OutT* __restrict__ C,
                      int M, int Mpad, int N, int K) {
  __shared__ _Float16 As[128][40];   // +8 f16 pad: kills 4-way bank conflicts
  __shared__ _Float16 Bs[128][40];

  const int tid   = threadIdx.x;
  const int lane  = tid & 31;
  const int wid   = tid >> 5;
  const int lhalf = lane >> 4;       // 0 | 1 (lanes 0-15 / 16-31)
  const int l15   = lane & 15;

  const int tm = blockIdx.x * 128;
  const int tn = blockIdx.y * 128;
  const int wm = (wid & 3) * 32;     // wave M offset in tile
  const int wn = (wid >> 2) * 64;    // wave N offset in tile

  v8f acc[2][4] = {};

  const int lrow = tid >> 1;         // 0..127 : tile row this thread loads
  const int lk   = (tid & 1) * 16;   // 0 | 16 : k-subchunk

  for (int k0 = 0; k0 < K; k0 += 32) {
    // ---- cooperative load A tile (f16-convert), clamp rows >= M to zero ----
    {
      H16 tmp;
      const int ra = tm + lrow;
      if (ra < M) {
        if constexpr (sizeof(InT) == 4) {
          const float* pa = (const float*)A + (size_t)ra * K + k0 + lk;
#pragma unroll
          for (int i = 0; i < 16; i += 4) {
            float4 f = *(const float4*)(pa + i);
            tmp.e[i + 0] = (_Float16)f.x; tmp.e[i + 1] = (_Float16)f.y;
            tmp.e[i + 2] = (_Float16)f.z; tmp.e[i + 3] = (_Float16)f.w;
          }
        } else {
          const _Float16* pa = (const _Float16*)A + (size_t)ra * K + k0 + lk;
          tmp.h[0] = *(const v8h*)(pa);
          tmp.h[1] = *(const v8h*)(pa + 8);
        }
      } else {
#pragma unroll
        for (int i = 0; i < 16; ++i) tmp.e[i] = (_Float16)0.0f;
      }
      *(v8h*)&As[lrow][lk]     = tmp.h[0];
      *(v8h*)&As[lrow][lk + 8] = tmp.h[1];
    }
    // ---- cooperative load W tile (rows are N, contiguous over K) ----
    {
      H16 tmp;
      const float* pw = W + (size_t)(tn + lrow) * K + k0 + lk;
#pragma unroll
      for (int i = 0; i < 16; i += 4) {
        float4 f = *(const float4*)(pw + i);
        tmp.e[i + 0] = (_Float16)f.x; tmp.e[i + 1] = (_Float16)f.y;
        tmp.e[i + 2] = (_Float16)f.z; tmp.e[i + 3] = (_Float16)f.w;
      }
      *(v8h*)&Bs[lrow][lk]     = tmp.h[0];
      *(v8h*)&Bs[lrow][lk + 8] = tmp.h[1];
    }
    __syncthreads();

    // ---- WMMA: A-frag per lane: row m=l15, K = lhalf*16 + i (contiguous 16) ----
    V16 a[2];
#pragma unroll
    for (int i = 0; i < 2; ++i) {
      const int m = wm + i * 16 + l15;
      a[i].h[0] = *(const v8h*)&As[m][lhalf * 16];
      a[i].h[1] = *(const v8h*)&As[m][lhalf * 16 + 8];
    }
#pragma unroll
    for (int j = 0; j < 4; ++j) {
      const int n = wn + j * 16 + l15;
      V16 bb;           // B-frag: col n=l15, K = lhalf*16 + i (contiguous 16)
      bb.h[0] = *(const v8h*)&Bs[n][lhalf * 16];
      bb.h[1] = *(const v8h*)&Bs[n][lhalf * 16 + 8];
#pragma unroll
      for (int i = 0; i < 2; ++i)
        acc[i][j] = wmma_f16f32(a[i].v, bb.v, acc[i][j]);
    }
    __syncthreads();
  }

  // ---- store: C layout: lane col n=l15, VGPR r -> row r + 8*lhalf ----
#pragma unroll
  for (int i = 0; i < 2; ++i) {
    const int rbase = tm + wm + i * 16 + lhalf * 8;
#pragma unroll
    for (int j = 0; j < 4; ++j) {
      const int col = tn + wn + j * 16 + l15;
      const float bv = bias[col];
#pragma unroll
      for (int r = 0; r < 8; ++r) {
        const int row = rbase + r;
        if (row < Mpad) {
          const float v = (row < M) ? (acc[i][j][r] + bv) : 0.0f;
          C[(size_t)row * N + col] = (OutT)v;
        }
      }
    }
  }
}

// ---------------------------------------------------------------------------
// Flash cross-attention. grid = B*H*(L/128) = 512 workgroups, 256 thr (8 waves)
// Wave owns 16 query rows; Q fragments live in registers (E=128 = 4 k-chunks).
// S loop: 16 chunks of 64 prototypes; K chunk async-copied s-major into LDS
// (GLOBAL_LOAD_ASYNC_TO_LDS_B128, no VGPR round-trip), V chunk e-major
// (transposed through VGPRs) so both B-fragments are contiguous 32B reads.
// Online softmax: row stats per lane cover rows r+8*lhalf; N-reduction is a
// __shfl_xor butterfly over the 16-lane half (matches C-matrix layout).
// P is re-staged through per-wave LDS to convert C-layout -> A-layout.
// ---------------------------------------------------------------------------
#define S_VALID 1000

__global__ __launch_bounds__(256)
void attn_flash_kernel(const _Float16* __restrict__ Qh,
                       const _Float16* __restrict__ Kh,
                       const _Float16* __restrict__ Vh,
                       _Float16* __restrict__ Oh) {
  __shared__ _Float16 Ks[64][136];       // [s][e]  (+8 pad)
  __shared__ _Float16 Vt[128][72];       // [e][s]  (+8 pad)
  __shared__ _Float16 Ps[8][16][72];     // per-wave P staging [m][s]

  const int tid   = threadIdx.x;
  const int lane  = tid & 31;
  const int wid   = tid >> 5;
  const int lhalf = lane >> 4;
  const int l15   = lane & 15;

  const int lb = blockIdx.x & 7;          // L block
  const int h  = (blockIdx.x >> 3) & 7;   // head
  const int b  = blockIdx.x >> 6;         // batch

  const int qrow0   = b * 1024 + lb * 128 + wid * 16; // row in [8192]
  const int colbase = h * 128;                        // head cols in 1024-wide

  // ---- Q fragments (A-layout), held in registers for the whole S loop ----
  V16 aq[4];
  {
    const _Float16* qrow = Qh + (size_t)(qrow0 + l15) * 1024 + colbase;
    const int kb = lhalf * 16;
#pragma unroll
    for (int kc = 0; kc < 4; ++kc) {
      aq[kc].h[0] = *(const v8h*)(qrow + kc * 32 + kb);
      aq[kc].h[1] = *(const v8h*)(qrow + kc * 32 + kb + 8);
    }
  }

  v8f o[8] = {};                         // O accum: 16 rows x 128 E
  float rowM[8], rowL[8];
#pragma unroll
  for (int r = 0; r < 8; ++r) { rowM[r] = -3.0e38f; rowL[r] = 0.0f; }

  const float scale = 0.08838834764831845f; // 1/sqrt(128)

  for (int sc = 0; sc < 16; ++sc) {
    const int s0 = sc * 64;
    // ---- K: async global->LDS copy (ASYNCcnt), V: transpose via VGPRs ----
#pragma unroll
    for (int it = 0; it < 4; ++it) {
      const int idx = it * 256 + tid;     // 1024 groups of 8 f16
      const int rr = idx >> 4;            // s within chunk (0..63)
      const int cc = (idx & 15) * 8;      // e (0..120)
      // generic LDS pointer: addr[31:0] is the raw LDS offset (ISA 10.2)
      const unsigned ldsdst = (unsigned)(size_t)&Ks[rr][cc];
      const unsigned long long gsrc =
          (unsigned long long)(size_t)(Kh + (size_t)(s0 + rr) * 1024 + colbase + cc);
      asm volatile("global_load_async_to_lds_b128 %0, %1, off"
                   :: "v"(ldsdst), "v"(gsrc) : "memory");
      v8h dv = *(const v8h*)(Vh + (size_t)(s0 + rr) * 1024 + colbase + cc);
#pragma unroll
      for (int j = 0; j < 8; ++j) Vt[cc + j][rr] = dv[j];
    }
    asm volatile("s_wait_asynccnt 0" ::: "memory");  // drain this wave's copies
    __syncthreads();                                  // publish LDS to all waves

    // ---- scores: 4 S-tiles of 16, K-dim E=128 = 4 wmma steps ----
    v8f sacc[4];
#pragma unroll
    for (int t = 0; t < 4; ++t) {
      v8f c = {};
      const int srow = t * 16 + l15;      // B-frag col = s
#pragma unroll
      for (int kc = 0; kc < 4; ++kc) {
        V16 bk;
        bk.h[0] = *(const v8h*)&Ks[srow][kc * 32 + lhalf * 16];
        bk.h[1] = *(const v8h*)&Ks[srow][kc * 32 + lhalf * 16 + 8];
        c = wmma_f16f32(aq[kc].v, bk.v, c);
      }
      sacc[t] = c;
    }
    // ---- scale + mask padded prototypes (column index uniform per lane) ----
#pragma unroll
    for (int t = 0; t < 4; ++t) {
      const int colg = s0 + t * 16 + l15;
      if (colg >= S_VALID) {
#pragma unroll
        for (int r = 0; r < 8; ++r) sacc[t][r] = -3.0e38f;
      } else {
#pragma unroll
        for (int r = 0; r < 8; ++r) sacc[t][r] *= scale;
      }
    }
    // ---- online softmax: chunk row max (reduce tiles, then 16-lane half) ----
    float cm[8];
#pragma unroll
    for (int r = 0; r < 8; ++r)
      cm[r] = fmaxf(fmaxf(sacc[0][r], sacc[1][r]), fmaxf(sacc[2][r], sacc[3][r]));
#pragma unroll
    for (int d = 1; d < 16; d <<= 1)
#pragma unroll
      for (int r = 0; r < 8; ++r) cm[r] = fmaxf(cm[r], __shfl_xor(cm[r], d, 32));

    float alpha[8];
#pragma unroll
    for (int r = 0; r < 8; ++r) {
      const float nm = fmaxf(rowM[r], cm[r]);
      alpha[r] = __expf(rowM[r] - nm);
      rowM[r] = nm;
    }

    // ---- P = exp(s - M): accumulate row sums, stage to LDS as f16 ----
    float rs[8];
#pragma unroll
    for (int r = 0; r < 8; ++r) rs[r] = 0.0f;
#pragma unroll
    for (int t = 0; t < 4; ++t) {
#pragma unroll
      for (int r = 0; r < 8; ++r) {
        const float p = __expf(sacc[t][r] - rowM[r]);
        rs[r] += p;
        Ps[wid][r + lhalf * 8][t * 16 + l15] = (_Float16)p;
      }
    }
#pragma unroll
    for (int d = 1; d < 16; d <<= 1)
#pragma unroll
      for (int r = 0; r < 8; ++r) rs[r] += __shfl_xor(rs[r], d, 32);
#pragma unroll
    for (int r = 0; r < 8; ++r) rowL[r] = rowL[r] * alpha[r] + rs[r];

    // ---- rescale O accumulators (same row<->VGPR mapping as scores) ----
#pragma unroll
    for (int et = 0; et < 8; ++et)
#pragma unroll
      for (int r = 0; r < 8; ++r) o[et][r] *= alpha[r];

    // ---- O += P(16x64) x V(64x16 per e-tile): 2 wmma k-steps of 32 ----
#pragma unroll
    for (int kc = 0; kc < 2; ++kc) {
      V16 pa;                              // A-frag of P from staging
      pa.h[0] = *(const v8h*)&Ps[wid][l15][kc * 32 + lhalf * 16];
      pa.h[1] = *(const v8h*)&Ps[wid][l15][kc * 32 + lhalf * 16 + 8];
#pragma unroll
      for (int et = 0; et < 8; ++et) {
        V16 bv;                            // B-frag of V (e-major -> contiguous)
        bv.h[0] = *(const v8h*)&Vt[et * 16 + l15][kc * 32 + lhalf * 16];
        bv.h[1] = *(const v8h*)&Vt[et * 16 + l15][kc * 32 + lhalf * 16 + 8];
        o[et] = wmma_f16f32(pa.v, bv.v, o[et]);
      }
    }
    __syncthreads();
  }

  // ---- normalize and store f16 attention output ----
#pragma unroll
  for (int et = 0; et < 8; ++et) {
#pragma unroll
    for (int r = 0; r < 8; ++r) {
      const float v = o[et][r] / rowL[r];
      Oh[(size_t)(qrow0 + r + lhalf * 8) * 1024 + colbase + et * 16 + l15] =
          (_Float16)v;
    }
  }
}

// ---------------------------------------------------------------------------
// Launch: Qproj -> Kproj -> Vproj -> flash-attn -> Oproj
// Workspace (f16): Qh[8192x1024] | Kh[1024x1024] | Vh[1024x1024] | Oh[8192x1024]
// = 36 MB total, fully written before read.
// ---------------------------------------------------------------------------
extern "C" void kernel_launch(void* const* d_in, const int* in_sizes, int n_in,
                              void* d_out, int out_size, void* d_ws, size_t ws_size,
                              hipStream_t stream) {
  const float* tgt = (const float*)d_in[0];   // [8,1024,1024] -> [8192,1024]
  const float* src = (const float*)d_in[1];   // [1000,4096]
  const float* val = (const float*)d_in[2];   // [1000,4096]
  const float* Wq  = (const float*)d_in[3];   // [1024,1024]
  const float* bq  = (const float*)d_in[4];
  const float* Wk  = (const float*)d_in[5];   // [1024,4096]
  const float* bk  = (const float*)d_in[6];
  const float* Wv  = (const float*)d_in[7];   // [1024,4096]
  const float* bv  = (const float*)d_in[8];
  const float* Wo  = (const float*)d_in[9];   // [4096,1024]
  const float* bo  = (const float*)d_in[10];
  float* out = (float*)d_out;                 // [8192,4096]

  _Float16* Qh = (_Float16*)d_ws;
  _Float16* Kh = Qh + (size_t)8192 * 1024;
  _Float16* Vh = Kh + (size_t)1024 * 1024;
  _Float16* Oh = Vh + (size_t)1024 * 1024;

  const dim3 blk(256);
  // Q = tgt @ Wq^T + bq            [8192,1024]
  gemm_bias_kernel<float, _Float16><<<dim3(64, 8), blk, 0, stream>>>(
      tgt, Wq, bq, Qh, 8192, 8192, 1024, 1024);
  // K = src @ Wk^T + bk            [1000->1024, 1024], pad rows zeroed
  gemm_bias_kernel<float, _Float16><<<dim3(8, 8), blk, 0, stream>>>(
      src, Wk, bk, Kh, 1000, 1024, 1024, 4096);
  // V = val @ Wv^T + bv            [1000->1024, 1024], pad rows zeroed
  gemm_bias_kernel<float, _Float16><<<dim3(8, 8), blk, 0, stream>>>(
      val, Wv, bv, Vh, 1000, 1024, 1024, 4096);
  // flash cross-attention          Oh[8192,1024]
  attn_flash_kernel<<<dim3(512), blk, 0, stream>>>(Qh, Kh, Vh, Oh);
  // out = Oh @ Wo^T + bo           [8192,4096] f32
  gemm_bias_kernel<_Float16, float><<<dim3(64, 32), blk, 0, stream>>>(
      Oh, Wo, bo, out, 8192, 8192, 4096, 1024);
}
